// MambaFusion_17214228922834
// MI455X (gfx1250) — compile-verified, matrix operand order checked by reading, not run on hardware
//
#include <hip/hip_runtime.h>
#include <hip/hip_bf16.h>

// ============================================================================
// MambaFusion forward for MI455X (gfx1250, wave32, WMMA).
//
// Dense layers: operands are pre-converted f32 -> bf16 by a tiny elementwise
// kernel, then the WMMA GEMM loads bf16 fragments DIRECTLY into the operand
// registers: one fragment = two 16-byte runs = 2x global_load_b128, no VALU
// between load and v_wmma_f32_16x16x32_bf16. This halves GEMM-side L2 traffic
// (operands are re-read tilesM/tilesN times) and removes the cvt chains +
// WMMA->VALU hazard NOPs seen when converting in-loop. Each wave computes NT
// (1/2/4) 16x16 C tiles so one A fragment feeds NT back-to-back WMMAs.
//
// Workspace requirement: ~134 MB. Input flattening assumption: setup_inputs()
// dict insertion order, params flattened recursively = 218 tensors.
// ============================================================================

typedef __bf16 bf16_t;
typedef __attribute__((ext_vector_type(16))) __bf16        v16bf;
typedef __attribute__((ext_vector_type(8)))  __bf16        v8bf;
typedef __attribute__((ext_vector_type(8)))  float         v8f;
typedef __attribute__((ext_vector_type(4)))  unsigned int  v4u;

#define WAVE 32

enum { ACT_NONE = 0, ACT_GELU = 1, ACT_SILU = 2, ACT_SOFTPLUS = 3 };

// ---------------------------------------------------------------- math helpers
__device__ __forceinline__ float gelu_f(float x) {
    float x3 = x * x * x;
    return 0.5f * x * (1.0f + tanhf(0.7978845608028654f * (x + 0.044715f * x3)));
}
__device__ __forceinline__ float silu_f(float x) {
    return x / (1.0f + __expf(-x));
}
__device__ __forceinline__ float softplus_f(float x) {
    return (x > 20.0f) ? x : log1pf(__expf(x));
}
__device__ __forceinline__ float apply_act(float v, int act) {
    if (act == ACT_GELU) return gelu_f(v);
    if (act == ACT_SILU) return silu_f(v);
    if (act == ACT_SOFTPLUS) return softplus_f(v);
    return v;
}

// ---------------------------------------------------------------- bf16 frags
__device__ __forceinline__ v16bf cat8(v8bf lo, v8bf hi) {
    return __builtin_shufflevector(lo, hi, 0, 1, 2, 3, 4, 5, 6, 7,
                                           8, 9, 10, 11, 12, 13, 14, 15);
}
// Fast fragment: rowp = base + row*ld + 8*half (bf16 elements); two contiguous
// 8-bf16 (16-byte) runs at kk and kk+16 -> 2x global_load_b128, no VALU.
__device__ __forceinline__ v16bf frag_ld(const bf16_t* __restrict__ rowp, int kk) {
    v8bf lo = *(const v8bf*)(rowp + kk);
    v8bf hi = *(const v8bf*)(rowp + kk + 16);
    return cat8(lo, hi);
}
// Clamped 8-bf16 load, zeroed with a bitwise AND mask (keeps the load
// unconditional -> batched). K % 8 == 0 at every call site.
__device__ __forceinline__ v8bf ld8_guard(const bf16_t* __restrict__ base, int k, int K) {
    int kc = (k + 8 <= K) ? k : (K - 8);
    unsigned msk = (k + 8 <= K) ? 0xFFFFFFFFu : 0u;
    v4u raw = *(const v4u*)(base + kc);
    raw &= msk;
    return __builtin_bit_cast(v8bf, raw);
}
__device__ __forceinline__ v16bf frag_ld_guard(const bf16_t* __restrict__ base, int half,
                                               int kk, int K) {
    int k0 = kk + half * 8;
    return cat8(ld8_guard(base, k0, K), ld8_guard(base, k0 + 16, K));
}

// ------------------------------------------------------- f32 -> bf16 convert
__global__ __launch_bounds__(256)
void cvt_bf16_kernel(const float* __restrict__ in, bf16_t* __restrict__ out,
                     long long n8)   // element count / 8
{
    long long i = blockIdx.x * (long long)blockDim.x + threadIdx.x;
    if (i >= n8) return;
    const float4* p = (const float4*)(in + i * 8);
    float4 a = p[0], b = p[1];
    v8bf r;
    r[0] = (bf16_t)a.x; r[1] = (bf16_t)a.y; r[2] = (bf16_t)a.z; r[3] = (bf16_t)a.w;
    r[4] = (bf16_t)b.x; r[5] = (bf16_t)b.y; r[6] = (bf16_t)b.z; r[7] = (bf16_t)b.w;
    *(v8bf*)(out + i * 8) = r;
}

// ---------------------------------------------------------------- WMMA GEMM
// C[M,N] = act(Abf[M,K] @ Wbf[N,K]^T + bias) (+ res[M,N]);  A/W in bf16.
// One wave computes NT horizontally-adjacent 16x16 C tiles (A frag reused).
// Fragment layouts per CDNA5 ISA 7.12.2 (wave32):
//   A 16x32 bf16: lane m = lane&15, half = lane>>4; elems = A[m][kk+8h..+7],
//                 A[m][kk+16+8h..+7]
//   B 32x16 bf16: lane n = lane&15, same K mapping (column-major per lane)
//   C 16x16 f32 : vgpr j -> m = j + 8*half, n = lane&15
// Host guarantees full M/N tiles whenever NT >= 2.
template<int NT>
__global__ __launch_bounds__(256)
__attribute__((amdgpu_waves_per_eu(2)))
void wmma_gemm_kernel(const bf16_t* __restrict__ A, int lda,
                      const bf16_t* __restrict__ W,           // (N,K) row-major
                      const float* __restrict__ bias,         // (N) or null
                      const float* __restrict__ res, int ldr, // (M,N) or null
                      float* __restrict__ C, int ldc,
                      int M, int N, int K, int act)
{
    int gwave = (int)((blockIdx.x * (size_t)blockDim.x + threadIdx.x) >> 5);
    int lane  = threadIdx.x & 31;
    int tilesN = (N + 16 * NT - 1) / (16 * NT);
    int tilesM = (M + 15) >> 4;
    if (gwave >= tilesM * tilesN) return;   // wave-uniform: EXEC stays all-ones
    int tm = gwave / tilesN;
    int tn = gwave % tilesN;
    int l15 = lane & 15, half = lane >> 4;
    int row = tm * 16 + l15;

    v8f zero8 = {};
    v8f acc[NT];
#pragma unroll
    for (int nt = 0; nt < NT; ++nt) acc[nt] = zero8;

    int cols[NT];
#pragma unroll
    for (int nt = 0; nt < NT; ++nt) cols[nt] = tn * 16 * NT + nt * 16 + l15;

    // wave-uniform full-tile test; always true for NT >= 2 (host contract)
    bool fast = true;
    if constexpr (NT == 1)
        fast = (tm * 16 + 16 <= M) && (tn * 16 + 16 <= N);

    if (fast) {
        const bf16_t* Arow = A + (size_t)row * lda + half * 8;
        const bf16_t* Wrow[NT];
#pragma unroll
        for (int nt = 0; nt < NT; ++nt)
            Wrow[nt] = W + (size_t)cols[nt] * K + half * 8;

        int kfull = K & ~31;
#pragma unroll 2
        for (int kk = 0; kk < kfull; kk += 32) {
            v16bf af = frag_ld(Arow, kk);
#pragma unroll
            for (int nt = 0; nt < NT; ++nt) {
                v16bf bf = frag_ld(Wrow[nt], kk);
                acc[nt] = __builtin_amdgcn_wmma_f32_16x16x32_bf16(
                    false, af, false, bf, (short)0, acc[nt], false, false);
            }
        }
        if (kfull < K) {            // K tail (e.g. dt projection, K=16)
            v16bf af = frag_ld_guard(A + (size_t)row * lda, half, kfull, K);
#pragma unroll
            for (int nt = 0; nt < NT; ++nt) {
                v16bf bf = frag_ld_guard(W + (size_t)cols[nt] * K, half, kfull, K);
                acc[nt] = __builtin_amdgcn_wmma_f32_16x16x32_bf16(
                    false, af, false, bf, (short)0, acc[nt], false, false);
            }
        }
    } else if constexpr (NT == 1) {
        // Edge tiles (only the final 16x4 head): clamped scalar gather.
        int rc = (row < M) ? row : (M - 1);
        int cc = (cols[0] < N) ? cols[0] : (N - 1);
        for (int kk = 0; kk < K; kk += 32) {
            v16bf af, bf;
#pragma unroll
            for (int e = 0; e < 16; ++e) {
                int pair = e >> 1;
                int k = kk + ((pair >> 2) << 4) + (half << 3) + ((pair & 3) << 1) + (e & 1);
                int kc = (k < K) ? k : (K - 1);
                bf16_t av = A[(size_t)rc * lda + kc];
                bf16_t bv = W[(size_t)cc * K   + kc];
                af[e] = (row < M && k < K)     ? av : (bf16_t)0.0f;
                bf[e] = (cols[0] < N && k < K) ? bv : (bf16_t)0.0f;
            }
            acc[0] = __builtin_amdgcn_wmma_f32_16x16x32_bf16(
                false, af, false, bf, (short)0, acc[0], false, false);
        }
    }

    // fused epilogue: bias -> activation -> residual -> store
#pragma unroll
    for (int nt = 0; nt < NT; ++nt) {
        int n = cols[nt];
#pragma unroll
        for (int j = 0; j < 8; ++j) {
            int m = tm * 16 + j + (half << 3);
            if (m < M && n < N) {
                float v = acc[nt][j];
                if (bias) v += bias[n];
                v = apply_act(v, act);
                if (res)  v += res[(size_t)m * ldr + n];
                C[(size_t)m * ldc + n] = v;
            }
        }
    }
}

// ---------------------------------------------------------------- LayerNorm
__global__ __launch_bounds__(256)
void ln_kernel(const float* __restrict__ x, const float* __restrict__ w,
               const float* __restrict__ b, float* __restrict__ out,
               int rows, int D, int fuse_gelu)
{
    int wv   = (int)((blockIdx.x * (size_t)blockDim.x + threadIdx.x) >> 5);
    int lane = threadIdx.x & 31;
    if (wv >= rows) return;
    const float* xr = x + (size_t)wv * D;
    float s = 0.0f;
    for (int i = lane; i < D; i += WAVE) s += xr[i];
#pragma unroll
    for (int o = 16; o > 0; o >>= 1) s += __shfl_xor(s, o, WAVE);
    float mean = s / (float)D;
    float var = 0.0f;
    for (int i = lane; i < D; i += WAVE) { float d = xr[i] - mean; var += d * d; }
#pragma unroll
    for (int o = 16; o > 0; o >>= 1) var += __shfl_xor(var, o, WAVE);
    float inv = rsqrtf(var / (float)D + 1e-5f);
    for (int i = lane; i < D; i += WAVE) {
        float y = (xr[i] - mean) * inv * w[i] + b[i];
        if (fuse_gelu) y = gelu_f(y);
        out[(size_t)wv * D + i] = y;
    }
}

// ------------------------------------------------- causal depthwise conv + SiLU
__global__ __launch_bounds__(256)
void conv_silu_kernel(const float* __restrict__ xz,   // (rows, 1024)
                      const float* __restrict__ cw,   // (512, 4)
                      const float* __restrict__ cb,   // (512)
                      float* __restrict__ xc,         // (rows, 512)
                      int rows, int L)
{
    size_t idx = blockIdx.x * (size_t)blockDim.x + threadIdx.x;
    if (idx >= (size_t)rows * 512) return;
    int d = (int)(idx & 511);
    size_t bt = idx >> 9;              // = b*L + t
    int t = (int)(bt % (size_t)L);
    float acc = cb[d];
#pragma unroll
    for (int j = 0; j < 4; ++j) {
        int tt = t - 3 + j;
        if (tt >= 0) acc += cw[d * 4 + j] * xz[(bt - 3 + j) * 1024 + d];
    }
    xc[idx] = silu_f(acc);
}

// ---------------------------------------------------------------- selective scan
// One block per batch element, 512 threads = channels; 16-state recurrence in
// registers, per-timestep B/C (32 floats) broadcast via LDS. Prefetch next
// timestep's operands (global_prefetch) while the recurrence chain computes.
__global__ __launch_bounds__(512)
void scan_kernel(const float* __restrict__ delta,  // (rows, 512)
                 const float* __restrict__ xdbl,   // (rows, 48)  [dt|B|C]
                 const float* __restrict__ xc,     // (rows, 512)
                 const float* __restrict__ xz,     // (rows, 1024) z at +512
                 const float* __restrict__ A_log,  // (512, 16)
                 const float* __restrict__ Dp,     // (512)
                 float* __restrict__ y,            // (rows, 512)
                 int L)
{
    int b = blockIdx.x;
    int d = threadIdx.x;
    __shared__ float Bs[16];
    __shared__ float Cs[16];
    float Ar[16];
#pragma unroll
    for (int s = 0; s < 16; ++s) Ar[s] = -__expf(A_log[d * 16 + s]);
    float h[16];
#pragma unroll
    for (int s = 0; s < 16; ++s) h[s] = 0.0f;
    float Dd = Dp[d];

    for (int t = 0; t < L; ++t) {
        size_t bt = (size_t)b * L + t;
        if (d < 32) {
            float v = xdbl[bt * 48 + 16 + d];
            if (d < 16) Bs[d] = v; else Cs[d - 16] = v;
        }
        __syncthreads();
        float dt = delta[bt * 512 + d];
        float xv = xc[bt * 512 + d];
        if (t + 1 < L) {   // hide next-step latency behind the 16-state chain
            __builtin_prefetch(delta + (bt + 1) * 512 + d, 0, 3);
            __builtin_prefetch(xc    + (bt + 1) * 512 + d, 0, 3);
            __builtin_prefetch(xz    + (bt + 1) * 1024 + 512 + d, 0, 3);
        }
        float yv = 0.0f;
#pragma unroll
        for (int s = 0; s < 16; ++s) {
            h[s] = __expf(dt * Ar[s]) * h[s] + (dt * xv) * Bs[s];
            yv  += h[s] * Cs[s];
        }
        float zv = xz[bt * 1024 + 512 + d];
        y[bt * 512 + d] = (yv + xv * Dd) * silu_f(zv);
        __syncthreads();
    }
}

// ---------------------------------------------------------------- cross-attention
// One wave per (b, head, query); lane = head-dim component (dh == 32).
__global__ __launch_bounds__(256)
void attn_kernel(const float* __restrict__ Q, const float* __restrict__ K,
                 const float* __restrict__ V, float* __restrict__ O,
                 int B, int Lq, int Lk,
                 const int* __restrict__ key_mask,   // (B,Lk) or null
                 const int* __restrict__ key_valid)  // (B)    or null
{
    int wv   = (int)((blockIdx.x * (size_t)blockDim.x + threadIdx.x) >> 5);
    int lane = threadIdx.x & 31;
    if (wv >= B * 8 * Lq) return;
    int q  = wv % Lq;
    int bh = wv / Lq;
    int h  = bh % 8;
    int b  = bh / 8;
    const float scale = 0.17677669529663687f;  // 1/sqrt(32)

    float qv = Q[((size_t)b * Lq + q) * 256 + h * 32 + lane];
    int nvalid = key_valid ? key_valid[b] : Lk;

    float m = -1e30f, l = 0.0f, acc = 0.0f;
    for (int kk = 0; kk < Lk; ++kk) {
        size_t koff = ((size_t)b * Lk + kk) * 256 + h * 32 + lane;
        if (kk + 8 < Lk) {            // keep K/V streams ahead of the ALU chain
            size_t poff = koff + (size_t)8 * 256;
            __builtin_prefetch(K + poff, 0, 3);
            __builtin_prefetch(V + poff, 0, 3);
        }
        float s = qv * K[koff];
#pragma unroll
        for (int o = 16; o > 0; o >>= 1) s += __shfl_xor(s, o, WAVE);
        bool valid = key_mask ? (key_mask[(size_t)b * Lk + kk] != 0) : (kk < nvalid);
        s = valid ? s * scale : -1e9f;
        float mn = fmaxf(m, s);
        float f = __expf(m - mn);
        float p = __expf(s - mn);
        l   = l * f + p;
        acc = acc * f + p * V[koff];
        m = mn;
    }
    O[((size_t)b * Lq + q) * 256 + h * 32 + lane] = acc / l;
}

// ---------------------------------------------------------------- masked mean
__global__ __launch_bounds__(256)
void mean_kernel(const float* __restrict__ x, float* __restrict__ out,
                 int B, int L,
                 const int* __restrict__ mask, const int* __restrict__ valid,
                 int col_off, int out_ld)
{
    int idx = (int)(blockIdx.x * (size_t)blockDim.x + threadIdx.x);
    if (idx >= B * 256) return;
    int d = idx & 255;
    int b = idx >> 8;
    int nv = valid ? valid[b] : L;
    float s = 0.0f, c = 0.0f;
    for (int t = 0; t < L; ++t) {
        bool v = mask ? (mask[(size_t)b * L + t] != 0) : (t < nv);
        if (v) { s += x[((size_t)b * L + t) * 256 + d]; c += 1.0f; }
    }
    out[(size_t)b * out_ld + col_off + d] = s / fmaxf(c, 1e-9f);
}

// ------------------------------------------------- audio valid-frame reduction
__global__ __launch_bounds__(256)
void audio_valid_kernel(const int* __restrict__ mask, int* __restrict__ valid,
                        int nsamp, int maxF)
{
    __shared__ int red[256];
    int b = blockIdx.x;
    long long s = 0;
    for (int i = threadIdx.x; i < nsamp; i += blockDim.x)
        s += mask[(size_t)b * nsamp + i];
    red[threadIdx.x] = (int)s;
    __syncthreads();
    for (int o = 128; o > 0; o >>= 1) {
        if ((int)threadIdx.x < o) red[threadIdx.x] += red[threadIdx.x + o];
        __syncthreads();
    }
    if (threadIdx.x == 0) {
        int f = red[0] / 320;                 // Wav2Vec2 stride 320
        if (f < 1) f = 1;
        if (f > maxF) f = maxF;
        valid[b] = f;
    }
}

// ============================================================================
// Host side
// ============================================================================
static inline int ceil_div(long long a, long long b) { return (int)((a + b - 1) / b); }

extern "C" void kernel_launch(void* const* d_in, const int* in_sizes, int n_in,
                              void* d_out, int out_size, void* d_ws, size_t ws_size,
                              hipStream_t stream)
{
    (void)in_sizes; (void)n_in; (void)out_size; (void)ws_size;

    constexpr int  Bn = 16, ST = 256, SA = 500, DM = 256, ENC = 768;
    constexpr int  DI = 512, FFd = 1024;
    constexpr long long MT = (long long)Bn * ST;   // 4096 text rows
    constexpr long long MA = (long long)Bn * SA;   // 8000 audio rows

    auto F = [&](int i) { return (const float*)d_in[i]; };
    auto I = [&](int i) { return (const int*)d_in[i]; };
    constexpr int IN_TEXT = 0, IN_AUDIO = 1, IN_TMASK = 2, IN_AMASK = 3;
    constexpr int P_TPROJ = 4, P_APROJ = 8;
    constexpr int P_TMAMBA = 12, P_AMAMBA = 80;
    constexpr int P_CROSS = 148;
    constexpr int P_CLF = 212;
    constexpr int L_LNW=0, L_LNB=1, L_INW=2, L_CONVW=3, L_CONVB=4, L_XPW=5,
                  L_DTW=6, L_DTB=7, L_ALOG=8, L_D=9, L_OUTW=10;
    constexpr int C_WQ=0,C_BQ=1,C_WK=2,C_BK=3,C_WV=4,C_BV=5,C_WO=6,C_BO=7,
                  C_LN1W=8,C_LN1B=9,C_FW1=10,C_FB1=11,C_FW2=12,C_FB2=13,
                  C_LN2W=14,C_LN2B=15;

    // ---- workspace layout (floats); total ~33.5M floats ~= 134 MB
    float* ws = (float*)d_ws;
    size_t off = 0;
    auto alloc = [&](size_t n) { float* p = ws + off; off += n; return p; };
    float* Tp0  = alloc(MT * DM);
    float* Tp1  = alloc(MT * DM);
    float* Ap0  = alloc(MA * DM);
    float* Ap1  = alloc(MA * DM);
    float* POOL = alloc((size_t)Bn * 2 * DM);
    float* H1   = alloc((size_t)Bn * DM);
    float* H1G  = alloc((size_t)Bn * DM);
    int*   AVALID = (int*)alloc(16);
    bf16_t* BFA = (bf16_t*)alloc(4096000);   // 8.192M bf16: max A operand (MA*FFd)
    bf16_t* BFW = (bf16_t*)alloc(131072);    // 262144 bf16: max W operand (1024x256)
    float* SCR  = alloc(23000000);
    // mamba-phase scratch
    float* LNB = SCR;
    float* XZ  = SCR + MA * DM;
    float* XC  = XZ  + MA * 2 * DI;
    float* XDB = XC  + MA * DI;
    float* DEL = XDB + MA * 48;
    float* YB  = DEL + MA * DI;
    // attention-phase scratch (same region, sequential phases)
    float* QB  = SCR;
    float* KB  = SCR + 1 * MA * DM;
    float* VB  = SCR + 2 * MA * DM;
    float* OB  = SCR + 3 * MA * DM;
    float* T1s = SCR + 4 * MA * DM;
    float* X1s = SCR + 5 * MA * DM;
    float* FFH = SCR + 6 * MA * DM;
    float* T2s = FFH + MA * FFd;

    auto cvt = [&](const float* src, bf16_t* dst, long long nelem) {
        long long n8 = nelem / 8;      // all operand sizes are multiples of 8
        cvt_bf16_kernel<<<dim3(ceil_div(n8, 256)), dim3(256), 0, stream>>>(src, dst, n8);
    };
    auto gemm = [&](const float* A, int lda, const float* W, const float* bias,
                    const float* res, int ldr, float* C, int ldc,
                    long long M, int N, int K, int act) {
        cvt(A, BFA, M * (long long)lda);
        cvt(W, BFW, (long long)N * K);
        // NT >= 2 requires full tiles: M % 16 == 0 and N % (16*NT) == 0
        int NT = 1;
        if (M % 16 == 0) NT = (N % 64 == 0) ? 4 : (N % 32 == 0) ? 2 : 1;
        long long tiles = (long long)ceil_div(M, 16) * ceil_div(N, 16 * NT);
        int blocks = ceil_div(tiles, 8);  // 8 waves / 256-thread block
        if (NT == 4)
            wmma_gemm_kernel<4><<<dim3(blocks), dim3(256), 0, stream>>>(
                BFA, lda, BFW, bias, res, ldr, C, ldc, (int)M, N, K, act);
        else if (NT == 2)
            wmma_gemm_kernel<2><<<dim3(blocks), dim3(256), 0, stream>>>(
                BFA, lda, BFW, bias, res, ldr, C, ldc, (int)M, N, K, act);
        else
            wmma_gemm_kernel<1><<<dim3(blocks), dim3(256), 0, stream>>>(
                BFA, lda, BFW, bias, res, ldr, C, ldc, (int)M, N, K, act);
    };
    auto ln = [&](const float* x, const float* w, const float* b, float* out,
                  long long rows, int D, int fuse_gelu) {
        int blocks = ceil_div(rows * 32, 256);
        ln_kernel<<<dim3(blocks), dim3(256), 0, stream>>>(x, w, b, out,
                                                          (int)rows, D, fuse_gelu);
    };

    // ---- audio frame mask
    audio_valid_kernel<<<dim3(Bn), dim3(256), 0, stream>>>(I(IN_AMASK), AVALID,
                                                           160000, SA);

    // ---- input projections: gelu(ln(x @ w.T + b))
    gemm(F(IN_TEXT), ENC, F(P_TPROJ + 0), F(P_TPROJ + 1), nullptr, 0,
         LNB, DM, MT, DM, ENC, ACT_NONE);
    ln(LNB, F(P_TPROJ + 2), F(P_TPROJ + 3), Tp0, MT, DM, 1);
    gemm(F(IN_AUDIO), ENC, F(P_APROJ + 0), F(P_APROJ + 1), nullptr, 0,
         LNB, DM, MA, DM, ENC, ACT_NONE);
    ln(LNB, F(P_APROJ + 2), F(P_APROJ + 3), Ap0, MA, DM, 1);

    // ---- mamba encoders
    auto run_mamba = [&](float* X, long long R, int L, int pbase) {
        for (int i = 0; i < 6; ++i) {
            int lb = pbase + 11 * i;
            ln(X, F(lb + L_LNW), F(lb + L_LNB), LNB, R, DM, 0);
            gemm(LNB, DM, F(lb + L_INW), nullptr, nullptr, 0,
                 XZ, 2 * DI, R, 2 * DI, DM, ACT_NONE);
            conv_silu_kernel<<<dim3(ceil_div(R * DI, 256)), dim3(256), 0, stream>>>(
                XZ, F(lb + L_CONVW), F(lb + L_CONVB), XC, (int)R, L);
            gemm(XC, DI, F(lb + L_XPW), nullptr, nullptr, 0,
                 XDB, 48, R, 48, DI, ACT_NONE);
            gemm(XDB, 48, F(lb + L_DTW), F(lb + L_DTB), nullptr, 0,
                 DEL, DI, R, DI, 16, ACT_SOFTPLUS);
            scan_kernel<<<dim3(Bn), dim3(DI), 0, stream>>>(
                DEL, XDB, XC, XZ, F(lb + L_ALOG), F(lb + L_D), YB, L);
            gemm(YB, DI, F(lb + L_OUTW), nullptr, X, DM,
                 X, DM, R, DM, DI, ACT_NONE);
        }
        ln(X, F(pbase + 66), F(pbase + 67), X, R, DM, 0);
    };
    run_mamba(Tp0, MT, ST, P_TMAMBA);
    run_mamba(Ap0, MA, SA, P_AMAMBA);

    // ---- cross-attention layers
    auto cross_dir = [&](const float* Xq, long long Rq, int Lq,
                         const float* Ykv, long long Rk, int Lk,
                         const int* kmask, const int* kvalid,
                         int pb, float* OUT) {
        gemm(Xq,  DM, F(pb + C_WQ), F(pb + C_BQ), nullptr, 0, QB, DM, Rq, DM, DM, ACT_NONE);
        gemm(Ykv, DM, F(pb + C_WK), F(pb + C_BK), nullptr, 0, KB, DM, Rk, DM, DM, ACT_NONE);
        gemm(Ykv, DM, F(pb + C_WV), F(pb + C_BV), nullptr, 0, VB, DM, Rk, DM, DM, ACT_NONE);
        attn_kernel<<<dim3(ceil_div((long long)Bn * 8 * Lq * 32, 256)),
                      dim3(256), 0, stream>>>(QB, KB, VB, OB, Bn, Lq, Lk, kmask, kvalid);
        gemm(OB, DM, F(pb + C_WO), F(pb + C_BO), Xq, DM, T1s, DM, Rq, DM, DM, ACT_NONE);
        ln(T1s, F(pb + C_LN1W), F(pb + C_LN1B), X1s, Rq, DM, 0);
        gemm(X1s, DM, F(pb + C_FW1), F(pb + C_FB1), nullptr, 0, FFH, FFd, Rq, FFd, DM, ACT_GELU);
        gemm(FFH, FFd, F(pb + C_FW2), F(pb + C_FB2), X1s, DM, T2s, DM, Rq, DM, FFd, ACT_NONE);
        ln(T2s, F(pb + C_LN2W), F(pb + C_LN2B), OUT, Rq, DM, 0);
    };

    float *tc = Tp0, *tn = Tp1, *ac = Ap0, *an = Ap1;
    for (int l = 0; l < 2; ++l) {
        int pb = P_CROSS + l * 32;
        cross_dir(tc, MT, ST, ac, MA, SA, nullptr, AVALID, pb, tn);
        cross_dir(ac, MA, SA, tc, MT, ST, I(IN_TMASK), nullptr, pb + 16, an);
        float* t = tc; tc = tn; tn = t;
        t = ac; ac = an; an = t;
    }

    // ---- masked mean pooling -> fused (B, 512)
    mean_kernel<<<dim3(ceil_div((long long)Bn * 256, 256)), dim3(256), 0, stream>>>(
        tc, POOL, Bn, ST, I(IN_TMASK), nullptr, 0, 2 * DM);
    mean_kernel<<<dim3(ceil_div((long long)Bn * 256, 256)), dim3(256), 0, stream>>>(
        ac, POOL, Bn, SA, nullptr, AVALID, DM, 2 * DM);

    // ---- classifier: gelu(ln(fused @ w1.T + b1)) @ w2.T + b2
    gemm(POOL, 2 * DM, F(P_CLF + 0), F(P_CLF + 1), nullptr, 0,
         H1, DM, Bn, DM, 2 * DM, ACT_NONE);
    ln(H1, F(P_CLF + 2), F(P_CLF + 3), H1G, Bn, DM, 1);
    gemm(H1G, DM, F(P_CLF + 4), F(P_CLF + 5), nullptr, 0,
         (float*)d_out, 4, Bn, 4, DM, ACT_NONE);
}
